// RNN_68247030334405
// MI455X (gfx1250) — matrix operand order repaired and checked by reading, not compiled, not run on hardware
//
#include <hip/hip_runtime.h>
#include <hip/hip_bf16.h>
#include <cstddef>

// ---------------------------------------------------------------------------
// Problem constants (from reference)
// ---------------------------------------------------------------------------
#define SEQ_LEN   64
#define BATCH     32
#define VOCAB     50257
#define VOCAB_PAD 50272          // next multiple of 16
#define HID       512
#define EMB       512

typedef __attribute__((ext_vector_type(16))) _Float16 v16h;
typedef __attribute__((ext_vector_type(8)))  _Float16 v8h;
typedef __attribute__((ext_vector_type(8)))  float    v8f;

#define WMMA_F32_F16(a, b, c) \
    __builtin_amdgcn_wmma_f32_16x16x32_f16(false, (a), false, (b), (short)0, (c), false, false)

// ---------------------------------------------------------------------------
// Per-lane fragment loaders (CDNA5 wave32 WMMA layouts, CDNA5 ISA)
//   A 16x32 f16:  lane<16 -> row M=lane,    K = {k0..k0+7, k0+16..k0+23}
//                 lane>=16 -> row M=lane-16, K = {k0+8..k0+15, k0+24..k0+31}
//   B 32x16 f16:  lane<16 -> col N=lane,    K = {k0..k0+7, k0+16..k0+23}
//                 lane>=16 -> col N=lane-16, K = {k0+8..k0+15, k0+24..k0+31}
// Caller pre-applies the half-select (hs) offsets, so here:
//   ldA(p): halves at p[0..7] and p[16..23]
//   ldB(p): halves at p[0..7] and p[16..23]
// ---------------------------------------------------------------------------
__device__ __forceinline__ v16h pack16(v8h lo, v8h hi) {
    v16h r;
#pragma unroll
    for (int i = 0; i < 8; ++i) { r[i] = lo[i]; r[i + 8] = hi[i]; }
    return r;
}
__device__ __forceinline__ v16h ldA(const _Float16* p) {
    return pack16(*(const v8h*)p, *(const v8h*)(p + 16));
}
__device__ __forceinline__ v16h ldB(const _Float16* p) {
    return pack16(*(const v8h*)p, *(const v8h*)(p + 16));
}

// ---------------------------------------------------------------------------
// Fused dual-matmul K-loop for one 16-wide output column group n0:
//   acc(m-tile0/1) += A1(32x512) x B1 + A2(32x512) x B2
// A1/A2 live in LDS (32x512 f16 row-major), B1/B2 in global (row = out neuron,
// 512 contiguous K halves per row -> serves as column-major K x N fragment).
// ---------------------------------------------------------------------------
__device__ __forceinline__ void gemm_pair(const _Float16* A1, const _Float16* A2,
                                          const _Float16* __restrict__ B1,
                                          const _Float16* __restrict__ B2,
                                          int n0, int lane, v8f& acc0, v8f& acc1) {
    const int hs = lane >> 4;      // half-wave select
    const int nl = lane & 15;
    const _Float16* a1p = A1 + nl * HID + hs * 8;
    const _Float16* a1q = A1 + (nl + 16) * HID + hs * 8;
    const _Float16* a2p = A2 + nl * HID + hs * 8;
    const _Float16* a2q = A2 + (nl + 16) * HID + hs * 8;
    const _Float16* b1p = B1 + (size_t)(n0 + nl) * HID + hs * 8;
    const _Float16* b2p = B2 + (size_t)(n0 + nl) * HID + hs * 8;
#pragma unroll 4
    for (int k0 = 0; k0 < HID; k0 += 32) {
        v16h bx  = ldB(b1p + k0);
        v16h bh  = ldB(b2p + k0);
        v16h ax0 = ldA(a1p + k0);
        v16h ax1 = ldA(a1q + k0);
        v16h ah0 = ldA(a2p + k0);
        v16h ah1 = ldA(a2q + k0);
        acc0 = WMMA_F32_F16(ax0, bx, acc0);
        acc0 = WMMA_F32_F16(ah0, bh, acc0);
        acc1 = WMMA_F32_F16(ax1, bx, acc1);
        acc1 = WMMA_F32_F16(ah1, bh, acc1);
    }
}

// One RNN layer: DST = tanh(A1 @ B1^T + A2 @ B2^T + biases). DST may alias A2.
// 512 threads = 16 waves; wave w owns column groups w*16 and (w+16)*16.
__device__ __forceinline__ void rnn_layer(const _Float16* A1, const _Float16* A2,
                                          const _Float16* __restrict__ B1,
                                          const _Float16* __restrict__ B2,
                                          const float* __restrict__ bias1,
                                          const float* __restrict__ bias2,
                                          _Float16* DST, int lane, int wave) {
    v8f accA0 = {}, accA1 = {}, accB0 = {}, accB1 = {};
    gemm_pair(A1, A2, B1, B2, wave * 16,        lane, accA0, accA1);
    gemm_pair(A1, A2, B1, B2, (wave + 16) * 16, lane, accB0, accB1);

    const int hs = lane >> 4;
    const int nl = lane & 15;
    const int nA = wave * 16 + nl;
    const int nB = (wave + 16) * 16 + nl;
    const float bA = (bias1 ? bias1[nA] : 0.0f) + (bias2 ? bias2[nA] : 0.0f);
    const float bB = (bias1 ? bias1[nB] : 0.0f) + (bias2 ? bias2[nB] : 0.0f);

    __syncthreads();               // all reads of A2 (old state) complete
#pragma unroll
    for (int r = 0; r < 8; ++r) {  // C/D layout: VGPR r -> M=r (lanes 0-15) / M=r+8
        const int m = r + hs * 8;
        DST[m * HID + nA]        = (_Float16)tanhf(accA0[r] + bA);
        DST[(m + 16) * HID + nA] = (_Float16)tanhf(accA1[r] + bA);
        DST[m * HID + nB]        = (_Float16)tanhf(accB0[r] + bB);
        DST[(m + 16) * HID + nB] = (_Float16)tanhf(accB1[r] + bB);
    }
    __syncthreads();               // new state visible to all waves
}

// ---------------------------------------------------------------------------
// Persistent recurrence kernel: one workgroup (512 thr = 16 wave32) walks all
// 64 timesteps; states live in *static* LDS (96 KB total — validated at
// compile time against the gfx1250 WGP limit of 320 KB). Emits the per-step
// selected row (batch row NUM_LAYERS-1 == 1) into hX for the decoder GEMM,
// and final h1 (fp32) into the tail of d_out.
// ---------------------------------------------------------------------------
__global__ void __launch_bounds__(512)
rnn_recurrence_kernel(const int* __restrict__ inputs,
                      const float* __restrict__ hidden,
                      const float* __restrict__ W_emb,
                      const _Float16* __restrict__ hWreg0,
                      const float* __restrict__ b_reg0,
                      const _Float16* __restrict__ hWrec0,
                      const float* __restrict__ b_rec0,
                      const _Float16* __restrict__ hWreg1,
                      const _Float16* __restrict__ hWrec1,
                      const float* __restrict__ b_rec1,
                      _Float16* __restrict__ hX,
                      float* __restrict__ h1f_out) {
    __shared__ _Float16 X [BATCH * HID];   // 32 KB : layer-0 input (embeddings)
    __shared__ _Float16 H0[BATCH * HID];   // 32 KB : layer-0 hidden state
    __shared__ _Float16 H1[BATCH * HID];   // 32 KB : layer-1 hidden state

    const int tid  = threadIdx.x;        // 0..511
    const int lane = tid & 31;
    const int wave = tid >> 5;           // 0..15

    // init states from `hidden` input (shape (2, B, H))
    for (int i = tid; i < BATCH * HID; i += 512) {
        H0[i] = (_Float16)hidden[i];
        H1[i] = (_Float16)hidden[BATCH * HID + i];
    }
    __syncthreads();

    for (int t = 0; t < SEQ_LEN; ++t) {
        // Embedding gather: X[b][e] = W_emb[inputs[t,b]][e]  (f32 -> f16)
        {
            const int b  = tid >> 4;            // 0..31
            const int e0 = (tid & 15) * 32;     // 0..480
            const int tok = inputs[t * BATCH + b];
            const float* src = W_emb + (size_t)tok * EMB + e0;
            _Float16* dst = X + b * HID + e0;
#pragma unroll
            for (int i = 0; i < 32; ++i) dst[i] = (_Float16)src[i];
        }
        __syncthreads();

        // layer 0: H0 = tanh(X @ Wreg0^T + b_reg0 + H0 @ Wrec0^T + b_rec0)
        rnn_layer(X, H0, hWreg0, hWrec0, b_reg0, b_rec0, H0, lane, wave);
        // layer 1: H1 = tanh(H0 @ Wreg1^T + H1 @ Wrec1^T + b_rec1)   (no b_reg1)
        rnn_layer(H0, H1, hWreg1, hWrec1, nullptr, b_rec1, H1, lane, wave);

        // decoder input row: batch row index (NUM_LAYERS-1)==1 of H1
        if (tid < HID) hX[t * HID + tid] = H1[1 * HID + tid];
        // (next-iter gather overwrites X only; X reads finished before the
        //  mid-layer barrier inside layer 0, so no extra barrier needed)
    }

    __syncthreads();
    for (int i = tid; i < BATCH * HID; i += 512)
        h1f_out[i] = (float)H1[i];
}

// ---------------------------------------------------------------------------
// Decoder GEMM: Z(64 x VOCAB_PAD) = hX(64x512) x hWdec^T, one wave per 16x16
// tile, K=512 -> 16 WMMA per tile. Epilogue fuses + b_dec and the batch
// broadcast store into logits (S, B, V).
// ---------------------------------------------------------------------------
#define DEC_NTILES ((VOCAB_PAD / 16) * (SEQ_LEN / 16))   // 3142 * 4 = 12568

__global__ void decoder_kernel(const _Float16* __restrict__ hX,
                               const _Float16* __restrict__ hWdec,
                               const float* __restrict__ b_dec,
                               float* __restrict__ out) {
    const int wid = blockIdx.x * (blockDim.x >> 5) + (threadIdx.x >> 5);
    if (wid >= DEC_NTILES) return;           // wave-uniform (EXEC stays all-1)
    const int lane  = threadIdx.x & 31;
    const int mtile = wid & 3;               // 4 row tiles (S=64)
    const int ntile = wid >> 2;              // VOCAB_PAD/16 col tiles
    const int hs = lane >> 4;
    const int nl = lane & 15;

    const _Float16* ap = hX + (size_t)(mtile * 16 + nl) * HID + hs * 8;
    const _Float16* bp = hWdec + (size_t)(ntile * 16 + nl) * HID + hs * 8;

    v8f acc = {};
#pragma unroll 4
    for (int k0 = 0; k0 < HID; k0 += 32) {
        v16h a = ldA(ap + k0);
        v16h b = ldB(bp + k0);
        acc = WMMA_F32_F16(a, b, acc);
    }

    const int v = ntile * 16 + nl;
    if (v < VOCAB) {
        const float bias = b_dec[v];
#pragma unroll
        for (int r = 0; r < 8; ++r) {
            const int t = mtile * 16 + r + hs * 8;
            const float val = acc[r] + bias;
            float* o = out + (size_t)t * BATCH * VOCAB + v;
            // source bug reproduced: broadcast the selected row across batch
            for (int b = 0; b < BATCH; ++b)
                o[(size_t)b * VOCAB] = val;
        }
    }
}

// ---------------------------------------------------------------------------
// Weight converters (fp32 -> f16); W_dec zero-padded to VOCAB_PAD rows.
// ---------------------------------------------------------------------------
__global__ void cvt_f32_to_f16(const float* __restrict__ s,
                               _Float16* __restrict__ d, int n) {
    const int i = blockIdx.x * blockDim.x + threadIdx.x;
    if (i < n) d[i] = (_Float16)s[i];
}

__global__ void cvt_wdec_kernel(const float* __restrict__ s,
                                _Float16* __restrict__ d) {
    const size_t i = (size_t)blockIdx.x * blockDim.x + threadIdx.x;
    if (i >= (size_t)VOCAB_PAD * HID) return;
    const size_t v = i >> 9;                 // /HID
    d[i] = (v < VOCAB) ? (_Float16)s[i] : (_Float16)0.0f;
}

// ---------------------------------------------------------------------------
// Host launch
// ---------------------------------------------------------------------------
extern "C" void kernel_launch(void* const* d_in, const int* in_sizes, int n_in,
                              void* d_out, int out_size, void* d_ws, size_t ws_size,
                              hipStream_t stream) {
    (void)in_sizes; (void)n_in; (void)out_size; (void)ws_size;

    const int*   inputs = (const int*)  d_in[0];
    const float* hidden = (const float*)d_in[1];
    const float* W_emb  = (const float*)d_in[2];
    const float* W_reg0 = (const float*)d_in[3];
    const float* b_reg0 = (const float*)d_in[4];
    const float* W_rec0 = (const float*)d_in[5];
    const float* b_rec0 = (const float*)d_in[6];
    const float* W_reg1 = (const float*)d_in[7];
    const float* W_rec1 = (const float*)d_in[8];
    const float* b_rec1 = (const float*)d_in[9];
    const float* W_dec  = (const float*)d_in[10];
    const float* b_dec  = (const float*)d_in[11];

    float* out = (float*)d_out;
    float* h1f_out = out + (size_t)SEQ_LEN * BATCH * VOCAB;

    // workspace layout (f16 buffers, all 16B-aligned)
    char* ws = (char*)d_ws;
    size_t o = 0;
    _Float16* hWreg0 = (_Float16*)(ws + o); o += (size_t)HID * EMB * 2;
    _Float16* hWrec0 = (_Float16*)(ws + o); o += (size_t)HID * HID * 2;
    _Float16* hWreg1 = (_Float16*)(ws + o); o += (size_t)HID * HID * 2;
    _Float16* hWrec1 = (_Float16*)(ws + o); o += (size_t)HID * HID * 2;
    _Float16* hWdec  = (_Float16*)(ws + o); o += (size_t)VOCAB_PAD * HID * 2;
    _Float16* hX     = (_Float16*)(ws + o); o += (size_t)SEQ_LEN * HID * 2;

    // 1) convert weights to f16
    const int nw = HID * HID;                         // 262144
    cvt_f32_to_f16<<<nw / 256, 256, 0, stream>>>(W_reg0, hWreg0, nw);
    cvt_f32_to_f16<<<nw / 256, 256, 0, stream>>>(W_rec0, hWrec0, nw);
    cvt_f32_to_f16<<<nw / 256, 256, 0, stream>>>(W_reg1, hWreg1, nw);
    cvt_f32_to_f16<<<nw / 256, 256, 0, stream>>>(W_rec1, hWrec1, nw);
    {
        const size_t ndec = (size_t)VOCAB_PAD * HID;  // 25,739,264
        const int blocks = (int)((ndec + 255) / 256);
        cvt_wdec_kernel<<<blocks, 256, 0, stream>>>(W_dec, hWdec);
    }

    // 2) serial recurrence: single persistent workgroup, 96KB static LDS
    rnn_recurrence_kernel<<<1, 512, 0, stream>>>(
        inputs, hidden, W_emb,
        hWreg0, b_reg0, hWrec0, b_rec0,
        hWreg1, hWrec1, b_rec1,
        hX, h1f_out);

    // 3) decoder GEMM + fused batch-broadcast store of logits
    {
        const int waves_per_block = 8;                 // 256 threads
        const int blocks = (DEC_NTILES + waves_per_block - 1) / waves_per_block;
        decoder_kernel<<<blocks, 256, 0, stream>>>(hX, hWdec, b_dec, out);
    }
}
